// MultiscaleTensorField_72662256713900
// MI455X (gfx1250) — compile-verified, matrix operand order checked by reading, not run on hardware
//
#include <hip/hip_runtime.h>

// ---------------- problem constants ----------------
#define NQ   8192
#define NS   32768
#define KNN  32
#define C    128
#define H    8
#define HD   16
#define FC0  64
#define NL   2
#define E0   (NQ * KNN)      // 262144 edges per scale
#define ETOT (2 * E0)        // 524288 edges total

typedef __attribute__((ext_vector_type(16))) __bf16 v16bf;
typedef __attribute__((ext_vector_type(8)))  __bf16 v8bf;
typedef __attribute__((ext_vector_type(8)))  float  v8f;

// ---------------- CDNA5 async load-to-LDS (guarded) ----------------
#if defined(__has_builtin)
#if __has_builtin(__builtin_amdgcn_global_load_async_to_lds_b128) && \
    __has_builtin(__builtin_amdgcn_s_wait_asynccnt)
#define USE_ASYNC_LDS 1
#endif
#endif
#ifndef USE_ASYNC_LDS
#define USE_ASYNC_LDS 0
#endif

typedef int v4i_vec __attribute__((__vector_size__(4 * sizeof(int))));
typedef __attribute__((address_space(1))) v4i_vec* g_v4i_ptr;
typedef __attribute__((address_space(3))) v4i_vec* l_v4i_ptr;

// copy 16 bytes global -> LDS; async (ASYNCcnt) when available
__device__ inline void copy16_g2l(const __bf16* g, __bf16* l) {
#if USE_ASYNC_LDS
  __builtin_amdgcn_global_load_async_to_lds_b128((g_v4i_ptr)g, (l_v4i_ptr)l, 0,
                                                 0);
#else
  *(v8bf*)l = *(const v8bf*)g;
#endif
}
__device__ inline void async_wait0() {
#if USE_ASYNC_LDS
  __builtin_amdgcn_s_wait_asynccnt(0);
#endif
}

// ---------------- WMMA helpers ----------------
__device__ inline v8f wmma_bf16(v16bf a, v16bf b, v8f c) {
  return __builtin_amdgcn_wmma_f32_16x16x32_bf16(false, a, false, b, (short)0, c,
                                                 false, false);
}

// Fragment loader for 16x32 bf16 tiles stored row-major ([row][k], stride in
// halfs).  Lane layout per CDNA5 ISA: lanes 0-15 row M=lane hold K = kb..kb+7
// and kb+16..kb+23 with kb=0; lanes 16-31 same rows with kb=8. Two b128 loads.
__device__ inline v16bf load_frag(const __bf16* base, int strideHalfs, int k0) {
  const int lane = threadIdx.x & 31;
  const int r    = lane & 15;
  const int kb   = (lane >> 4) << 3;
  const __bf16* p = base + (size_t)r * strideHalfs + k0 + kb;
  v8bf lo = *(const v8bf*)p;          // K = kb .. kb+7
  v8bf hi = *(const v8bf*)(p + 16);   // K = kb+16 .. kb+23
  v16bf out;
#pragma unroll
  for (int i = 0; i < 8; ++i) { out[i] = lo[i]; out[i + 8] = hi[i]; }
  return out;
}

__device__ inline float silu(float x) { return x / (1.f + __expf(-x)); }

// ---------------- converters ----------------
__global__ void cvt_kernel(const float* __restrict__ s, __bf16* __restrict__ d,
                           long long n) {
  long long i = (long long)blockIdx.x * 256 + threadIdx.x;
  if (i < n) d[i] = (__bf16)s[i];
}

// transpose+convert weights: src f32 [Lc][K][N] -> dst bf16 [Lc][N][K]
__global__ void cvt_t_kernel(const float* __restrict__ s, __bf16* __restrict__ d,
                             int Lc, int K, int N) {
  long long i = (long long)blockIdx.x * 256 + threadIdx.x;
  long long tot = (long long)Lc * K * N;
  if (i >= tot) return;
  int n = (int)(i % N);
  long long rem = i / N;
  int k = (int)(rem % K);
  int l = (int)(rem / K);
  d[((size_t)l * N + n) * K + k] = (__bf16)s[i];
}

// combT[l][128][96] = transposed [wev(64 rows); wtp(9 rows); zeros]
__global__ void build_combT_kernel(const float* __restrict__ wev,
                                   const float* __restrict__ wtp,
                                   __bf16* __restrict__ combT) {
  long long i = (long long)blockIdx.x * 256 + threadIdx.x;
  if (i >= (long long)NL * 128 * 96) return;
  int r = (int)(i % 96);          // K index
  long long rem = i / 96;
  int n = (int)(rem % 128);       // N index
  int l = (int)(rem / 128);
  float v = 0.f;
  if (r < 64)          v = wev[((long long)l * 64 + r) * 128 + n];
  else if (r - 64 < 9) v = wtp[((long long)l * 9 + (r - 64)) * 128 + n];
  combT[i] = (__bf16)v;
}

// ---------------- generic bf16 WMMA GEMM ----------------
// C[M,N] = act( A[M,K] @ B[K,N] + bias + addsrc ); B given transposed [N][K].
// block = 256 threads = 8 waves; tile 128(M) x 64(N); K staged 64 at a time,
// double-buffered via CDNA5 async global->LDS DMA (ASYNCcnt) when available.
__global__ __launch_bounds__(256) void gemm_kernel(
    const __bf16* __restrict__ A, int lda, const __bf16* __restrict__ BT,
    int M, int N, int Kdim, const float* __restrict__ addsrc,
    const float* __restrict__ bias, int act, float* __restrict__ outF,
    __bf16* __restrict__ outB) {
  __shared__ __bf16 As[2][128 * 64];  // [buf][row][k]
  __shared__ __bf16 Bs[2][64 * 64];   // [buf][n][k]
  const int tid  = threadIdx.x;
  const int lane = tid & 31;
  const int wave = tid >> 5;
  const int row0 = blockIdx.x * 128;
  const int col0 = blockIdx.y * 64;

  auto stage = [&](int buf, int k0) {
    // A: 128x64 halfs -> 32 halfs/thread (4x b128 async)
#pragma unroll
    for (int i = 0; i < 4; ++i) {
      int idx = tid * 8 + i * 2048;
      int r = idx >> 6, kk = idx & 63;
      copy16_g2l(&A[(size_t)(row0 + r) * lda + k0 + kk], &As[buf][r * 64 + kk]);
    }
    // B (already transposed): 64x64 halfs -> 16 halfs/thread (2x b128 async)
#pragma unroll
    for (int i = 0; i < 2; ++i) {
      int idx = tid * 8 + i * 2048;
      int n = idx >> 6, kk = idx & 63;
      copy16_g2l(&BT[(size_t)(col0 + n) * Kdim + k0 + kk],
                 &Bs[buf][n * 64 + kk]);
    }
  };

  v8f acc[4];
#pragma unroll
  for (int t = 0; t < 4; ++t) { v8f z = {}; acc[t] = z; }

  stage(0, 0);
  async_wait0();
  __syncthreads();

  int cur = 0;
  for (int k0 = 0; k0 < Kdim; k0 += 64) {
    const bool more = (k0 + 64 < Kdim);
    if (more) stage(cur ^ 1, k0 + 64);  // fire-and-forget next tile

#pragma unroll
    for (int kk0 = 0; kk0 < 64; kk0 += 32) {
      v16bf af = load_frag(&As[cur][(wave * 16) * 64], 64, kk0);
#pragma unroll
      for (int t = 0; t < 4; ++t) {
        v16bf bfb = load_frag(&Bs[cur][(t * 16) * 64], 64, kk0);
        acc[t] = wmma_bf16(af, bfb, acc[t]);
      }
    }
    if (more) async_wait0();
    __syncthreads();
    cur ^= 1;
  }

  const int mbase = (lane >> 4) << 3;
  const int nl    = lane & 15;
#pragma unroll
  for (int t = 0; t < 4; ++t) {
#pragma unroll
    for (int v = 0; v < 8; ++v) {
      int r = row0 + wave * 16 + mbase + v;
      int c = col0 + t * 16 + nl;
      float x = acc[t][v];
      if (bias)   x += bias[c];
      if (addsrc) x += addsrc[(size_t)r * N + c];
      if (act == 1) x = silu(x);
      if (outF) outF[(size_t)r * N + c] = x;
      if (outB) outB[(size_t)r * N + c] = (__bf16)x;
    }
  }
}

// ---------------- edge preprocessing ----------------
// Per edge: rel, dist, SH basis Y (padded to 32), cutoff wc, and
// es = silu(rbf(dist) @ pre_w + pre_b) via WMMA.  Block = 4 waves x 16 edges.
// prewT: bf16 [2][64(N)][64(K)], fragment-loaded directly from global.
__global__ __launch_bounds__(128) void edge_pre_kernel(
    const float* __restrict__ q_pos, const float* __restrict__ src_pos0,
    const float* __restrict__ src_pos1, const int* __restrict__ es0,
    const int* __restrict__ ed0, const int* __restrict__ es1,
    const int* __restrict__ ed1, const __bf16* __restrict__ prewT,
    const float* __restrict__ pre_b, __bf16* __restrict__ es_bf,
    __bf16* __restrict__ y_bf, float* __restrict__ wc_g) {
  __shared__ __bf16 rbfA[4][16 * 64];    // per-wave 16 edges x 64 centers
  __shared__ float  distS[4][16];
  const int tid = threadIdx.x, lane = tid & 31, wave = tid >> 5;
  const long long eblk  = (long long)blockIdx.x * 64;
  const int   scale  = (eblk >= E0) ? 1 : 0;
  const float radius = scale ? 0.5f : 0.2f;

  const long long eb = eblk + wave * 16;
  if (lane < 16) {
    long long e  = eb + lane;
    long long el = scale ? (e - E0) : e;
    int s = scale ? es1[el] : es0[el];
    int d = scale ? ed1[el] : ed0[el];
    const float* sp = scale ? src_pos1 : src_pos0;
    float rx = sp[s * 3 + 0] - q_pos[d * 3 + 0];
    float ry = sp[s * 3 + 1] - q_pos[d * 3 + 1];
    float rz = sp[s * 3 + 2] - q_pos[d * 3 + 2];
    float dist = sqrtf(rx * rx + ry * ry + rz * rz + 1e-12f);
    float inv = 1.f / dist;
    float x = rx * inv, y = ry * inv, z = rz * inv;
    const float s3 = 1.7320508f, s15 = 3.8729833f, s5 = 2.2360680f;
    float Y[9] = {1.f, s3 * x, s3 * y, s3 * z, s15 * x * y, s15 * y * z,
                  0.5f * s5 * (3.f * z * z - 1.f), s15 * x * z,
                  0.5f * s15 * (x * x - y * y)};
    __bf16* yp = y_bf + e * 32;
#pragma unroll
    for (int j = 0; j < 9; ++j) yp[j] = (__bf16)Y[j];
#pragma unroll
    for (int j = 9; j < 32; ++j) yp[j] = (__bf16)0.f;
    float q2 = dist / radius;
    float cc = fminf(fmaxf(1.f - q2 * q2, 0.f), 1.f);
    wc_g[e] = cc * cc;
    distS[wave][lane] = dist;
  }
  __syncthreads();
  {  // RBF: each lane fills 32 centers for one edge
    int   er = lane & 15;
    float d  = distS[wave][er];
    int   cb = (lane >> 4) * 32;
    float inv_s = 64.f / radius;  // 1/sigma, sigma = r/FC0
#pragma unroll
    for (int j = 0; j < 32; ++j) {
      int   ci  = cb + j;
      float cen = radius * (float)ci / 63.f;
      float tt  = (d - cen) * inv_s;
      rbfA[wave][er * 64 + ci] = (__bf16)__expf(-0.5f * tt * tt);
    }
  }
  __syncthreads();
  const int mbase = (lane >> 4) << 3, nl = lane & 15;
  v16bf a0 = load_frag(&rbfA[wave][0], 64, 0);
  v16bf a1 = load_frag(&rbfA[wave][0], 64, 32);
  const __bf16* pw = prewT + (size_t)scale * 64 * 64;
#pragma unroll
  for (int t = 0; t < 4; ++t) {
    v8f acc = {};
    acc = wmma_bf16(a0, load_frag(pw + (t * 16) * 64, 64, 0),  acc);
    acc = wmma_bf16(a1, load_frag(pw + (t * 16) * 64, 64, 32), acc);
#pragma unroll
    for (int v = 0; v < 8; ++v) {
      long long e = eb + mbase + v;
      int c = t * 16 + nl;
      es_bf[e * 64 + c] = (__bf16)silu(acc[v] + pre_b[scale * 64 + c]);
    }
  }
}

// ---------------- edge message kernel (per layer) ----------------
// ek = es@wek ; evtp = [es|Ypad]@[wev;wtp] ; gather-multiply epilogues:
//   a[e,h] = sum_d leaky(qd[dst]+ks[src]*ek) * walpha[h]  (head == col tile)
//   ve[e,c] = vs[src,c] * evtp[e,c]
// Weights fragment-loaded directly from global transposed layout (L2-resident).
__global__ __launch_bounds__(128) void edge_msg_kernel(
    const __bf16* __restrict__ es_bf, const __bf16* __restrict__ y_bf,
    const __bf16* __restrict__ ks_bf, const __bf16* __restrict__ vs_bf,
    const __bf16* __restrict__ qd_bf, const __bf16* __restrict__ wekT_l,
    const __bf16* __restrict__ combT_l, const float* __restrict__ wal_l,
    const int* __restrict__ es0, const int* __restrict__ ed0,
    const int* __restrict__ es1, const int* __restrict__ ed1,
    float* __restrict__ a_g, __bf16* __restrict__ ve_bf) {
  __shared__ int idxS[4][16], idxD[4][16];
  const int lane = threadIdx.x & 31, wave = threadIdx.x >> 5;
  const long long eb = (long long)blockIdx.x * 64 + wave * 16;

  if (lane < 16) {
    long long e = eb + lane;
    int scale = (e >= E0) ? 1 : 0;
    long long el = scale ? (e - E0) : e;
    idxS[wave][lane] = (scale ? (es1[el] + NS) : es0[el]);
    idxD[wave][lane] = (scale ? ed1[el] : ed0[el]);
  }
  __syncthreads();

  const int mbase = (lane >> 4) << 3, nl = lane & 15;

  // hoisted A fragments (shared across all 8 head tiles)
  v16bf ae0 = load_frag(es_bf + eb * 64, 64, 0);
  v16bf ae1 = load_frag(es_bf + eb * 64, 64, 32);
  v16bf ay  = load_frag(y_bf + eb * 32, 32, 0);

  // ---- pass 1: ek -> attention logits a[e,h] ----
#pragma unroll
  for (int t = 0; t < 8; ++t) {
    v8f acc = {};
    acc = wmma_bf16(ae0, load_frag(wekT_l + (t * 16) * 64, 64, 0),  acc);
    acc = wmma_bf16(ae1, load_frag(wekT_l + (t * 16) * 64, 64, 32), acc);
    float wv_ = wal_l[t * 16 + nl];
    float av[8];
#pragma unroll
    for (int v = 0; v < 8; ++v) {
      int m = mbase + v;
      int ch = t * 16 + nl;
      float ksv = (float)ks_bf[(size_t)idxS[wave][m] * 128 + ch];
      float qv  = (float)qd_bf[(size_t)idxD[wave][m] * 128 + ch];
      float x = qv + ksv * acc[v];
      x = (x > 0.f) ? x : 0.1f * x;  // leaky_relu 0.1
      av[v] = x * wv_;
    }
#pragma unroll
    for (int off = 1; off < 16; off <<= 1)
#pragma unroll
      for (int v = 0; v < 8; ++v) av[v] += __shfl_xor(av[v], off, 16);
    if (nl == 0) {
#pragma unroll
      for (int v = 0; v < 8; ++v)
        a_g[(eb + mbase + v) * 8 + t] = av[v];
    }
  }

  // ---- pass 2: evtp -> ve ----
#pragma unroll
  for (int t = 0; t < 8; ++t) {
    v8f acc = {};
    acc = wmma_bf16(ae0, load_frag(combT_l + (t * 16) * 96, 96, 0),  acc);
    acc = wmma_bf16(ae1, load_frag(combT_l + (t * 16) * 96, 96, 32), acc);
    acc = wmma_bf16(ay,  load_frag(combT_l + (t * 16) * 96, 96, 64), acc);
#pragma unroll
    for (int v = 0; v < 8; ++v) {
      int m = mbase + v;
      int ch = t * 16 + nl;
      float vsv = (float)vs_bf[(size_t)idxS[wave][m] * 128 + ch];
      ve_bf[(eb + m) * 128 + ch] = (__bf16)(vsv * acc[v]);
    }
  }
}

// ---------------- per-query softmax + aggregation ----------------
__global__ __launch_bounds__(128) void aggregate_kernel(
    const float* __restrict__ a_g, const float* __restrict__ wc_g,
    const __bf16* __restrict__ ve_bf, __bf16* __restrict__ agg_bf) {
  __shared__ float aE[64 * 8];
  __shared__ float wcs[64];
  const int q = blockIdx.x, tid = threadIdx.x;
  if (tid < 64) {
    long long e = (tid < 32) ? ((long long)q * 32 + tid)
                             : ((long long)E0 + (long long)q * 32 + (tid - 32));
    wcs[tid] = wc_g[e];
#pragma unroll
    for (int h = 0; h < 8; ++h) aE[tid * 8 + h] = a_g[e * 8 + h];
  }
  __syncthreads();
  if (tid < 8) {
    int h = tid;
    float m = -1e30f;
    for (int e = 0; e < 64; ++e) m = fmaxf(m, aE[e * 8 + h]);
    float den = 1e-8f;
    for (int e = 0; e < 64; ++e) {
      float p = wcs[e] * __expf(aE[e * 8 + h] - m);
      aE[e * 8 + h] = p;
      den += p;
    }
    float inv = 1.f / den;
    for (int e = 0; e < 64; ++e) aE[e * 8 + h] *= inv;
  }
  __syncthreads();
  {
    int c = tid, h = c >> 4;
    float acc = 0.f;
    for (int e = 0; e < 64; ++e) {
      long long eg = (e < 32) ? ((long long)q * 32 + e)
                              : ((long long)E0 + (long long)q * 32 + (e - 32));
      acc += aE[e * 8 + h] * (float)ve_bf[eg * 128 + c];
    }
    agg_bf[(size_t)q * 128 + c] = (__bf16)acc;
  }
}

// ---------------- host orchestration ----------------
extern "C" void kernel_launch(void* const* d_in, const int* in_sizes, int n_in,
                              void* d_out, int out_size, void* d_ws,
                              size_t ws_size, hipStream_t stream) {
  const float* q_pos    = (const float*)d_in[0];
  const float* src_pos0 = (const float*)d_in[1];
  const float* src_pos1 = (const float*)d_in[2];
  const float* q_feat   = (const float*)d_in[3];
  const float* src_f0   = (const float*)d_in[4];
  const float* src_f1   = (const float*)d_in[5];
  const float* pre_w    = (const float*)d_in[6];
  const float* pre_b    = (const float*)d_in[7];
  const float* wq   = (const float*)d_in[8];
  const float* wk   = (const float*)d_in[9];
  const float* wv   = (const float*)d_in[10];
  const float* wek  = (const float*)d_in[11];
  const float* wev  = (const float*)d_in[12];
  const float* wtp  = (const float*)d_in[13];
  const float* wal  = (const float*)d_in[14];
  const float* wo   = (const float*)d_in[15];
  const float* bo   = (const float*)d_in[16];
  const float* wskp = (const float*)d_in[17];
  const float* w1   = (const float*)d_in[18];
  const float* b1   = (const float*)d_in[19];
  const float* w2   = (const float*)d_in[20];
  const int* e_s0 = (const int*)d_in[22];
  const int* e_d0 = (const int*)d_in[23];
  const int* e_s1 = (const int*)d_in[24];
  const int* e_d1 = (const int*)d_in[25];

  size_t off = 0;
  auto alloc = [&](size_t bytes) -> void* {
    void* p = (char*)d_ws + off;
    off += (bytes + 255) & ~(size_t)255;
    return p;
  };
  __bf16* sfeat_bf = (__bf16*)alloc((size_t)2 * NS * C * 2);
  __bf16* dst_bf   = (__bf16*)alloc((size_t)NQ * C * 2);
  __bf16* es_bf    = (__bf16*)alloc((size_t)ETOT * 64 * 2);
  __bf16* y_bf     = (__bf16*)alloc((size_t)ETOT * 32 * 2);
  float*  wc_g     = (float*)alloc((size_t)ETOT * 4);
  __bf16* qd_bf    = (__bf16*)alloc((size_t)NQ * C * 2);
  __bf16* ks_bf    = (__bf16*)alloc((size_t)2 * NS * C * 2);
  __bf16* vs_bf    = (__bf16*)alloc((size_t)2 * NS * C * 2);
  float*  a_g      = (float*)alloc((size_t)ETOT * 8 * 4);
  __bf16* ve_bf    = (__bf16*)alloc((size_t)ETOT * C * 2);
  __bf16* agg_bf   = (__bf16*)alloc((size_t)NQ * C * 2);
  float*  t1_f     = (float*)alloc((size_t)NQ * C * 4);
  float*  out1_f   = (float*)alloc((size_t)NQ * C * 4);
  __bf16* out1_b   = (__bf16*)alloc((size_t)NQ * C * 2);
  __bf16* h_bf     = (__bf16*)alloc((size_t)NQ * 512 * 2);
  // transposed bf16 weights [N][K]
  __bf16* wq_t   = (__bf16*)alloc((size_t)NL * C * C * 2);
  __bf16* wk_t   = (__bf16*)alloc((size_t)NL * C * C * 2);
  __bf16* wv_t   = (__bf16*)alloc((size_t)NL * C * C * 2);
  __bf16* wo_t   = (__bf16*)alloc((size_t)NL * C * C * 2);
  __bf16* wsk_t  = (__bf16*)alloc((size_t)NL * C * C * 2);
  __bf16* wek_t  = (__bf16*)alloc((size_t)NL * 128 * 64 * 2);
  __bf16* combT  = (__bf16*)alloc((size_t)NL * 128 * 96 * 2);
  __bf16* w1_t   = (__bf16*)alloc((size_t)NL * 512 * 128 * 2);
  __bf16* w2_t   = (__bf16*)alloc((size_t)NL * 128 * 512 * 2);
  __bf16* prewT  = (__bf16*)alloc((size_t)2 * 64 * 64 * 2);

  auto cvt = [&](const float* s, __bf16* d, long long n) {
    cvt_kernel<<<(unsigned)((n + 255) / 256), 256, 0, stream>>>(s, d, n);
  };
  auto cvtT = [&](const float* s, __bf16* d, int Lc, int K, int N) {
    long long n = (long long)Lc * K * N;
    cvt_t_kernel<<<(unsigned)((n + 255) / 256), 256, 0, stream>>>(s, d, Lc, K, N);
  };
  auto gemm = [&](const __bf16* A, int lda, const __bf16* BT, int M, int N,
                  int Kd, const float* addsrc, const float* bias, int act,
                  float* oF, __bf16* oB) {
    dim3 grid((unsigned)(M / 128), (unsigned)(N / 64));
    gemm_kernel<<<grid, 256, 0, stream>>>(A, lda, BT, M, N, Kd, addsrc, bias,
                                          act, oF, oB);
  };

  // one-time conversions / weight transposes
  cvt(src_f0, sfeat_bf, (long long)NS * C);
  cvt(src_f1, sfeat_bf + (size_t)NS * C, (long long)NS * C);
  cvt(q_feat, dst_bf, (long long)NQ * C);
  cvtT(wq, wq_t, NL, C, C);
  cvtT(wk, wk_t, NL, C, C);
  cvtT(wv, wv_t, NL, C, C);
  cvtT(wo, wo_t, NL, C, C);
  cvtT(wskp, wsk_t, NL, C, C);
  cvtT(wek, wek_t, NL, 64, C);
  cvtT(w1, w1_t, NL, C, 512);
  cvtT(w2, w2_t, NL, 512, C);
  cvtT(pre_w, prewT, 2, 64, 64);
  build_combT_kernel<<<(NL * 128 * 96 + 255) / 256, 256, 0, stream>>>(wev, wtp,
                                                                      combT);
  edge_pre_kernel<<<ETOT / 64, 128, 0, stream>>>(
      q_pos, src_pos0, src_pos1, e_s0, e_d0, e_s1, e_d1, prewT, pre_b, es_bf,
      y_bf, wc_g);

  for (int l = 0; l < NL; ++l) {
    gemm(dst_bf, C, wq_t + (size_t)l * C * C, NQ, C, C, nullptr, nullptr, 0,
         nullptr, qd_bf);
    gemm(sfeat_bf, C, wk_t + (size_t)l * C * C, 2 * NS, C, C, nullptr, nullptr,
         0, nullptr, ks_bf);
    gemm(sfeat_bf, C, wv_t + (size_t)l * C * C, 2 * NS, C, C, nullptr, nullptr,
         0, nullptr, vs_bf);
    edge_msg_kernel<<<ETOT / 64, 128, 0, stream>>>(
        es_bf, y_bf, ks_bf, vs_bf, qd_bf, wek_t + (size_t)l * 128 * 64,
        combT + (size_t)l * 128 * 96, wal + (size_t)l * H * HD, e_s0, e_d0,
        e_s1, e_d1, a_g, ve_bf);
    aggregate_kernel<<<NQ, 128, 0, stream>>>(a_g, wc_g, ve_bf, agg_bf);
    gemm(agg_bf, C, wo_t + (size_t)l * C * C, NQ, C, C, nullptr,
         bo + (size_t)l * C, 0, t1_f, nullptr);
    gemm(dst_bf, C, wsk_t + (size_t)l * C * C, NQ, C, C, t1_f, nullptr, 0,
         out1_f, out1_b);
    gemm(out1_b, C, w1_t + (size_t)l * 512 * 128, NQ, 512, C, nullptr,
         b1 + (size_t)l * 512, 1, nullptr, h_bf);
    gemm(h_bf, 512, w2_t + (size_t)l * 128 * 512, NQ, C, 512, out1_f, nullptr,
         0, (l == NL - 1) ? (float*)d_out : nullptr, dst_bf);
  }
  (void)in_sizes; (void)n_in; (void)out_size; (void)ws_size;
}
